// ConvEGNN_29686813950281
// MI455X (gfx1250) — compile-verified, matrix operand order checked by reading, not run on hardware
//
#include <hip/hip_runtime.h>
#include <hip/hip_bf16.h>

#define NNODES 50000
#define NEDGES 800000

typedef __attribute__((ext_vector_type(16))) __bf16        bf16x16;
typedef __attribute__((ext_vector_type(8)))  float         f32x8;
typedef __attribute__((ext_vector_type(8)))  unsigned int  u32x8;

__device__ __forceinline__ unsigned short f2bf(float f) {
  unsigned int u = __float_as_uint(f);
  u += 0x7FFFu + ((u >> 16) & 1u);          // round-to-nearest-even
  return (unsigned short)(u >> 16);
}
__device__ __forceinline__ float bf2f(unsigned short s) {
  return __uint_as_float(((unsigned int)s) << 16);
}
__device__ __forceinline__ float silu_f(float v) { return v / (1.0f + __expf(-v)); }
__device__ __forceinline__ float sigm_f(float v) { return 1.0f / (1.0f + __expf(-v)); }

// K offset inside a 32-wide chunk for fragment register v (ISA 7.12.2, 16-bit A layout)
__device__ __forceinline__ int frag_k(int lane, int v) {
  return ((lane >> 4) << 3) + ((v >> 2) << 4) + ((v & 3) << 1);
}

// A fragment: 16 x 32 bf16 chunk from an LDS row-major tile (ld in halfwords)
__device__ __forceinline__ bf16x16 load_a_frag(const unsigned short* tile, int ld, int kc, int lane) {
  const int row = lane & 15;
  u32x8 r;
#pragma unroll
  for (int v = 0; v < 8; ++v) {
    const int k = kc * 32 + frag_k(lane, v);
    r[v] = *(const unsigned int*)(tile + row * ld + k);   // even k -> dword aligned
  }
  return __builtin_bit_cast(bf16x16, r);
}

// B fragment: pre-swizzled in workspace; one contiguous 32B per lane
__device__ __forceinline__ bf16x16 load_b_frag(const u32x8* __restrict__ wf, int kc, int nt, int lane) {
  return __builtin_bit_cast(bf16x16, wf[(kc * 8 + nt) * 32 + lane]);
}

// ---------------- prep: f32 weights -> bf16 WMMA-B-fragment layout ----------------
__global__ void egnn_prep_kernel(const float* __restrict__ w_e1, const float* __restrict__ w_e2,
                                 const float* __restrict__ w_h1, const float* __restrict__ w_h2,
                                 unsigned int* __restrict__ wf_e1, unsigned int* __restrict__ wf_e2,
                                 unsigned int* __restrict__ wf_h1, unsigned int* __restrict__ wf_h2) {
  int t = blockIdx.x * blockDim.x + threadIdx.x;
  const float* W; unsigned int* D; int K; int local;
  if      (t < 18432)         { W = w_e1; D = wf_e1; K = 257; local = t; }          // 9 kc
  else if (t < 18432 + 8192)  { W = w_e2; D = wf_e2; K = 128; local = t - 18432; }  // 4 kc
  else if (t < 26624 + 16384) { W = w_h1; D = wf_h1; K = 256; local = t - 26624; }  // 8 kc
  else if (t < 51200)         { W = w_h2; D = wf_h2; K = 128; local = t - 43008; }  // 4 kc
  else return;
  const int v = local & 7, lane = (local >> 3) & 31, nt = (local >> 8) & 7, kc = local >> 11;
  const int k   = kc * 32 + ((lane >> 4) << 3) + ((v >> 2) << 4) + ((v & 3) << 1);
  const int col = nt * 16 + (lane & 15);
  const float a = (k     < K) ? W[k * 128 + col]       : 0.0f;
  const float b = (k + 1 < K) ? W[(k + 1) * 128 + col] : 0.0f;
  D[local] = (unsigned int)f2bf(a) | ((unsigned int)f2bf(b) << 16);
}

__global__ void egnn_zero_kernel(float* __restrict__ p, int n) {
  int t = blockIdx.x * blockDim.x + threadIdx.x;
  if (t < n) p[t] = 0.0f;
}

// ---------------- edge MLP + gate + scatter ----------------
__global__ __launch_bounds__(128) void egnn_edge_kernel(
    const float* __restrict__ h, const float* __restrict__ x, const int* __restrict__ edges,
    const u32x8* __restrict__ wf_e1, const u32x8* __restrict__ wf_e2,
    const float* __restrict__ b_e1, const float* __restrict__ b_e2,
    const float* __restrict__ w_inf, const float* __restrict__ b_inf,
    float* __restrict__ m_i) {
  __shared__ unsigned short stage[4][16 * 288];   // gathered [h_s|h_d|dist|pad] bf16
  __shared__ unsigned short a2[4][16 * 128];      // layer-1 activations bf16
  __shared__ float gbuf[4][16];

  const int lane = threadIdx.x & 31;
  const int wave = threadIdx.x >> 5;
  const int colq = lane & 15;
  const int ebase = (blockIdx.x * 4 + wave) * 16;
  unsigned short* st = stage[wave];
  unsigned short* aw = a2[wave];

  // ---- gather/stage 16 edges x 288 bf16 features ----
  for (int r = 0; r < 16; ++r) {
    const int e = ebase + r;
    const int s = edges[2 * e], d = edges[2 * e + 1];
    const float dx = x[3 * s] - x[3 * d], dy = x[3 * s + 1] - x[3 * d + 1], dz = x[3 * s + 2] - x[3 * d + 2];
    const float dist = sqrtf(dx * dx + dy * dy + dz * dz);
#pragma unroll
    for (int i = lane; i < 288; i += 32) {
      float v;
      if      (i < 128)  v = h[s * 128 + i];
      else if (i < 256)  v = h[d * 128 + (i - 128)];
      else if (i == 256) v = dist;
      else               v = 0.0f;
      st[r * 288 + i] = f2bf(v);
    }
  }
  __builtin_amdgcn_wave_barrier();

  // ---- layer 1: [16x288] @ [288x128] (+bias, SiLU) -> a2 ----
#pragma unroll 1
  for (int nt = 0; nt < 8; ++nt) {
    f32x8 acc;
    const float bias = b_e1[nt * 16 + colq];
#pragma unroll
    for (int v = 0; v < 8; ++v) acc[v] = bias;
#pragma unroll
    for (int kc = 0; kc < 9; ++kc) {
      if (kc + 1 < 9) __builtin_prefetch(&wf_e1[((kc + 1) * 8 + nt) * 32 + lane], 0, 3);
      bf16x16 a = load_a_frag(st, 288, kc, lane);
      bf16x16 b = load_b_frag(wf_e1, kc, nt, lane);
      acc = __builtin_amdgcn_wmma_f32_16x16x32_bf16(false, a, false, b, (short)0, acc, false, false);
    }
#pragma unroll
    for (int v = 0; v < 8; ++v) {
      const int row = v + ((lane >> 4) << 3);
      aw[row * 128 + nt * 16 + colq] = f2bf(silu_f(acc[v]));
    }
  }
  __builtin_amdgcn_wave_barrier();

  // ---- layer 2: [16x128] @ [128x128] (+bias, SiLU) = m_ij; fused gate dot ----
  unsigned short* mb = st;   // reuse stage region for m_ij bf16 (16x128)
  f32x8 dot;
#pragma unroll
  for (int v = 0; v < 8; ++v) dot[v] = 0.0f;
#pragma unroll 1
  for (int nt = 0; nt < 8; ++nt) {
    f32x8 acc;
    const float bias = b_e2[nt * 16 + colq];
#pragma unroll
    for (int v = 0; v < 8; ++v) acc[v] = bias;
#pragma unroll
    for (int kc = 0; kc < 4; ++kc) {
      bf16x16 a = load_a_frag(aw, 128, kc, lane);
      bf16x16 b = load_b_frag(wf_e2, kc, nt, lane);
      acc = __builtin_amdgcn_wmma_f32_16x16x32_bf16(false, a, false, b, (short)0, acc, false, false);
    }
    const float wv = w_inf[nt * 16 + colq];
#pragma unroll
    for (int v = 0; v < 8; ++v) {
      const float m = silu_f(acc[v]);
      dot[v] += m * wv;
      const int row = v + ((lane >> 4) << 3);
      mb[row * 128 + nt * 16 + colq] = f2bf(m);
    }
  }
  // reduce gate dot across the 16 lanes of each half-wave (rows 0-7 / 8-15)
  const float binf = b_inf[0];
#pragma unroll
  for (int v = 0; v < 8; ++v) {
    float s = dot[v];
    s += __shfl_xor(s, 1, 32);
    s += __shfl_xor(s, 2, 32);
    s += __shfl_xor(s, 4, 32);
    s += __shfl_xor(s, 8, 32);
    if (colq == 0) gbuf[wave][v + ((lane >> 4) << 3)] = sigm_f(s + binf);
  }
  __builtin_amdgcn_wave_barrier();

  // ---- scatter e_ij * m_ij into m_i[e_st] ----
  for (int r = 0; r < 16; ++r) {
    const int e = ebase + r;
    const int s = edges[2 * e];
    const float g = gbuf[wave][r];
#pragma unroll
    for (int c = lane; c < 128; c += 32) {
      unsafeAtomicAdd(&m_i[s * 128 + c], g * bf2f(mb[r * 128 + c]));
    }
  }
}

// ---------------- node update MLP (in-place over d_out which holds m_i) ----------------
__global__ __launch_bounds__(128) void egnn_node_kernel(
    const float* __restrict__ h,
    const u32x8* __restrict__ wf_h1, const u32x8* __restrict__ wf_h2,
    const float* __restrict__ b_h1, const float* __restrict__ b_h2,
    float* __restrict__ hio) {
  __shared__ unsigned short stage[4][16 * 256];
  __shared__ unsigned short a2[4][16 * 128];

  const int lane = threadIdx.x & 31;
  const int wave = threadIdx.x >> 5;
  const int colq = lane & 15;
  const int nbase = (blockIdx.x * 4 + wave) * 16;
  if (nbase >= NNODES) return;
  unsigned short* st = stage[wave];
  unsigned short* aw = a2[wave];

  // ---- stage [h | m_i] 16 x 256 bf16 (read m_i rows before overwriting them) ----
  for (int r = 0; r < 16; ++r) {
    const int node = nbase + r;
    const bool ok = node < NNODES;
#pragma unroll
    for (int i = lane; i < 256; i += 32) {
      float v = 0.0f;
      if (ok) v = (i < 128) ? h[node * 128 + i] : hio[node * 128 + (i - 128)];
      st[r * 256 + i] = f2bf(v);
    }
  }
  __builtin_amdgcn_wave_barrier();

  // ---- layer 1: [16x256] @ [256x128] (+bias, SiLU) ----
#pragma unroll 1
  for (int nt = 0; nt < 8; ++nt) {
    f32x8 acc;
    const float bias = b_h1[nt * 16 + colq];
#pragma unroll
    for (int v = 0; v < 8; ++v) acc[v] = bias;
#pragma unroll
    for (int kc = 0; kc < 8; ++kc) {
      if (kc + 1 < 8) __builtin_prefetch(&wf_h1[((kc + 1) * 8 + nt) * 32 + lane], 0, 3);
      bf16x16 a = load_a_frag(st, 256, kc, lane);
      bf16x16 b = load_b_frag(wf_h1, kc, nt, lane);
      acc = __builtin_amdgcn_wmma_f32_16x16x32_bf16(false, a, false, b, (short)0, acc, false, false);
    }
#pragma unroll
    for (int v = 0; v < 8; ++v) {
      const int row = v + ((lane >> 4) << 3);
      aw[row * 128 + nt * 16 + colq] = f2bf(silu_f(acc[v]));
    }
  }
  __builtin_amdgcn_wave_barrier();

  // ---- layer 2: [16x128] @ [128x128] + bias + residual -> h_new ----
#pragma unroll 1
  for (int nt = 0; nt < 8; ++nt) {
    f32x8 acc;
    const float bias = b_h2[nt * 16 + colq];
#pragma unroll
    for (int v = 0; v < 8; ++v) acc[v] = bias;
#pragma unroll
    for (int kc = 0; kc < 4; ++kc) {
      bf16x16 a = load_a_frag(aw, 128, kc, lane);
      bf16x16 b = load_b_frag(wf_h2, kc, nt, lane);
      acc = __builtin_amdgcn_wmma_f32_16x16x32_bf16(false, a, false, b, (short)0, acc, false, false);
    }
#pragma unroll
    for (int v = 0; v < 8; ++v) {
      const int row = v + ((lane >> 4) << 3);
      const int node = nbase + row;
      const int col = nt * 16 + colq;
      if (node < NNODES) hio[node * 128 + col] = acc[v] + h[node * 128 + col];
    }
  }
}

extern "C" void kernel_launch(void* const* d_in, const int* in_sizes, int n_in,
                              void* d_out, int out_size, void* d_ws, size_t ws_size,
                              hipStream_t stream) {
  const float* h     = (const float*)d_in[0];
  const float* x     = (const float*)d_in[1];
  const int*   edges = (const int*)  d_in[2];
  const float* w_e1  = (const float*)d_in[3];
  const float* b_e1  = (const float*)d_in[4];
  const float* w_e2  = (const float*)d_in[5];
  const float* b_e2  = (const float*)d_in[6];
  const float* w_inf = (const float*)d_in[7];
  const float* b_inf = (const float*)d_in[8];
  const float* w_h1  = (const float*)d_in[9];
  const float* b_h1  = (const float*)d_in[10];
  const float* w_h2  = (const float*)d_in[11];
  const float* b_h2  = (const float*)d_in[12];
  float* out = (float*)d_out;

  // workspace: swizzled bf16 weight fragments (dword counts)
  unsigned int* wf_e1 = (unsigned int*)d_ws;     // 9*8*32*8  = 18432
  unsigned int* wf_e2 = wf_e1 + 18432;           // 4*8*32*8  =  8192
  unsigned int* wf_h1 = wf_e2 + 8192;            // 8*8*32*8  = 16384
  unsigned int* wf_h2 = wf_h1 + 16384;           // 4*8*32*8  =  8192

  egnn_prep_kernel<<<(51200 + 255) / 256, 256, 0, stream>>>(
      w_e1, w_e2, w_h1, w_h2, wf_e1, wf_e2, wf_h1, wf_h2);

  egnn_zero_kernel<<<(NNODES * 128 + 255) / 256, 256, 0, stream>>>(out, NNODES * 128);

  egnn_edge_kernel<<<NEDGES / 64, 128, 0, stream>>>(
      h, x, edges, (const u32x8*)wf_e1, (const u32x8*)wf_e2,
      b_e1, b_e2, w_inf, b_inf, out);

  egnn_node_kernel<<<(NNODES + 63) / 64, 128, 0, stream>>>(
      h, (const u32x8*)wf_h1, (const u32x8*)wf_h2, b_h1, b_h2, out);
}
